// MaskedCrossAttention_15212774162932
// MI455X (gfx1250) — matrix operand order repaired and checked by reading, not compiled, not run on hardware
//
#include <hip/hip_runtime.h>
#include <math.h>

// ---------------------------------------------------------------------------
// MaskedCrossAttention for MI455X (gfx1250, wave32, WMMA).
// bf16 data path, f32 WMMA accumulation (v_wmma_f32_16x16x32_bf16).
// Dense masked attention per 16-token tile; mask uses -FLT_MAX so the
// "no media yet" rows degenerate to a uniform softmax exactly like the ref.
// GEMMs: one wave computes a 32x64 C tile (8 WMMA accumulators) to double
// arithmetic intensity vs 16x64 blocking.
// ---------------------------------------------------------------------------

#define Bb     2
#define TTXT   4096
#define DIMC   1024
#define JTOK   512          // T_MED * N_MED
#define HEADS  8
#define DH     64
#define INNER  512          // HEADS * DH

typedef __bf16 bf16;
typedef __attribute__((ext_vector_type(16))) __bf16 v16bf;
typedef __attribute__((ext_vector_type(8)))  __bf16 v8bf;
typedef __attribute__((ext_vector_type(4)))  __bf16 v4bf;
typedef __attribute__((ext_vector_type(8)))  float  v8f;

union ABReg { v16bf v; v8bf h[2]; };

__device__ __forceinline__ v8bf load8(const bf16* p) { return *(const v8bf*)p; }

__device__ __forceinline__ v8f wmma_bf16(v16bf a, v16bf b, v8f c) {
  // D = A(16x32) * B(32x16) + C, f32 accumulate
  return __builtin_amdgcn_wmma_f32_16x16x32_bf16(false, a, false, b,
                                                 (short)0, c, false, false);
}

// --------------------------- LayerNorm -> bf16 -----------------------------
__global__ void ln_to_bf16_kernel(const float* __restrict__ x,
                                  const float* __restrict__ scale,
                                  const float* __restrict__ bias,
                                  bf16* __restrict__ xn) {
  const int row  = blockIdx.x;             // B*TTXT rows
  const int t    = threadIdx.x;            // 256 threads, 4 elems each
  const float* xr = x + (size_t)row * DIMC;
  float4 v = *(const float4*)(xr + t * 4);
  float s  = v.x + v.y + v.z + v.w;
  float ss = v.x*v.x + v.y*v.y + v.z*v.z + v.w*v.w;
  for (int off = 16; off; off >>= 1) {
    s  += __shfl_xor(s,  off);
    ss += __shfl_xor(ss, off);
  }
  __shared__ float rs[8], rss[8];
  const int wave = t >> 5, lane = t & 31;
  if (lane == 0) { rs[wave] = s; rss[wave] = ss; }
  __syncthreads();
  if (t == 0) {
    float a = 0.f, c = 0.f;
    for (int i = 0; i < 8; ++i) { a += rs[i]; c += rss[i]; }
    rs[0] = a; rss[0] = c;
  }
  __syncthreads();
  const float mean = rs[0] * (1.0f / DIMC);
  const float var  = rss[0] * (1.0f / DIMC) - mean * mean;
  const float inv  = rsqrtf(var + 1e-6f);
  float4 sc = *(const float4*)(scale + t * 4);
  float4 bi = *(const float4*)(bias  + t * 4);
  v4bf o;
  o[0] = (bf16)((v.x - mean) * inv * sc.x + bi.x);
  o[1] = (bf16)((v.y - mean) * inv * sc.y + bi.y);
  o[2] = (bf16)((v.z - mean) * inv * sc.z + bi.z);
  o[3] = (bf16)((v.w - mean) * inv * sc.w + bi.w);
  *(v4bf*)(xn + (size_t)row * DIMC + t * 4) = o;
}

// --------------------------- f32 -> bf16 copy ------------------------------
__global__ void f32_to_bf16_kernel(const float* __restrict__ src,
                                   bf16* __restrict__ dst, int n) {
  int i = blockIdx.x * blockDim.x + threadIdx.x;
  if (i < n) dst[i] = (bf16)src[i];
}

// ------------------- text_time = inclusive cumsum(bool) --------------------
// media_locations is a jax bool array -> 1 byte per element on device.
__global__ void text_time_kernel(const unsigned char* __restrict__ loc,
                                 int* __restrict__ tt) {
  const int b = blockIdx.x, lane = threadIdx.x;   // one wave32 per batch row
  const unsigned char* L = loc + (size_t)b * TTXT;
  int* T = tt + (size_t)b * TTXT;
  int carry = 0;
  for (int c = 0; c < TTXT; c += 32) {
    int x = L[c + lane] ? 1 : 0;
    for (int off = 1; off < 32; off <<= 1) {
      int y = __shfl_up(x, off);
      if (lane >= off) x += y;
    }
    T[c + lane] = carry + x;
    carry += __shfl(x, 31);
  }
}

// ------------- pack f32 row-major weight [K,N] into B-operand --------------
// Packed block (kt,nt) = 1KB: elem (lane,e) = W[kt*32 + (lane>>4)*8 + (e&7) + (e>>3)*16,
//                                             nt*16 + (lane&15)] * scl
// block index = nt*(K/32) + kt  (k-tiles contiguous for streaming)
__global__ void pack_w_kernel(const float* __restrict__ w, bf16* __restrict__ wp,
                              int K, int N, float scl, int n) {
  int i = blockIdx.x * blockDim.x + threadIdx.x;
  if (i >= n) return;
  const int e = i & 15, lane = (i >> 4) & 31, blk = i >> 9;
  const int kTiles = K >> 5;
  const int kt = blk % kTiles, nt = blk / kTiles;
  const int k = kt * 32 + ((lane >> 4) << 3) + (e & 7) + ((e >> 3) << 4);
  const int col = nt * 16 + (lane & 15);
  wp[i] = (bf16)(w[(size_t)k * N + col] * scl);
}

// ---------------- generic bf16 WMMA GEMM: C = A @ Bpacked ------------------
// A: bf16 row-major [M,K]; Bp: packed B-operand.
// One wave computes a 32x64 C tile: 2 A operands x 4 B operands -> 8 WMMAs
// per 32-deep k-step (6 x 32B operand loads per 8 WMMAs).
template <typename OutT>
__global__ void gemm_bf16_kernel(const bf16* __restrict__ A,
                                 const bf16* __restrict__ Bp,
                                 OutT* __restrict__ C,
                                 int M, int N, int K) {
  const int lane   = threadIdx.x & 31;
  const int waveId = blockIdx.x * (blockDim.x >> 5) + (threadIdx.x >> 5);
  const int ntg    = N >> 6;                       // 64-col groups
  const int mt     = waveId / ntg;                 // 32-row tiles
  const int ng     = waveId % ntg;
  if (mt >= (M >> 5)) return;                      // whole wave exits together
  const int kTiles = K >> 5;

  v8f acc[2][4] = {};
  // A operand addressing (ISA 7.12.2): row = lane&15, K-half = (lane>>4)*8
  const bf16* arow0 = A + (size_t)(mt * 32 + (lane & 15)) * K + ((lane >> 4) << 3);
  const bf16* arow1 = arow0 + (size_t)16 * K;
  for (int kt = 0; kt < kTiles; ++kt) {
    ABReg a0, a1;
    a0.h[0] = load8(arow0 + kt * 32);
    a0.h[1] = load8(arow0 + kt * 32 + 16);
    a1.h[0] = load8(arow1 + kt * 32);
    a1.h[1] = load8(arow1 + kt * 32 + 16);
    if (kt + 2 < kTiles) {                         // gfx1250 global_prefetch_b8
      __builtin_prefetch(arow0 + (kt + 2) * 32, 0, 3);
      __builtin_prefetch(arow1 + (kt + 2) * 32, 0, 3);
    }
#pragma unroll
    for (int t = 0; t < 4; ++t) {
      const bf16* bblk = Bp + ((size_t)((ng * 4 + t) * kTiles + kt)) * 512 + lane * 16;
      ABReg bm;
      bm.h[0] = load8(bblk);
      bm.h[1] = load8(bblk + 8);
      acc[0][t] = wmma_bf16(a0.v, bm.v, acc[0][t]);
      acc[1][t] = wmma_bf16(a1.v, bm.v, acc[1][t]);
    }
  }
  // D layout: col = lane&15 (+16*tile), row = r + (lane>>4)*8
#pragma unroll
  for (int half_m = 0; half_m < 2; ++half_m) {
#pragma unroll
    for (int t = 0; t < 4; ++t) {
      const int col = (ng * 4 + t) * 16 + (lane & 15);
#pragma unroll
      for (int r = 0; r < 8; ++r) {
        const int row = mt * 32 + half_m * 16 + r + ((lane >> 4) << 3);
        C[(size_t)row * N + col] = (OutT)acc[half_m][t][r];
      }
    }
  }
}

// -------- repack K (from kv buffer) into attention B-operand (K^T) ---------
// Per (b,h): blocks (jt 0..31) x (kt 0..1); elem: d = kt*32+(lane>>4)*8+(e&7)+(e>>3)*16,
// tok = jt*16+(lane&15); value = kv[b*512+tok, h*64+d]
__global__ void pack_k_kernel(const bf16* __restrict__ kv,
                              bf16* __restrict__ kop, int n) {
  int i = blockIdx.x * blockDim.x + threadIdx.x;
  if (i >= n) return;
  const int e = i & 15, lane = (i >> 4) & 31, blk = i >> 9;
  const int kt = blk & 1;
  const int jt = (blk >> 1) & 31;
  const int bh = blk >> 6;
  const int b = bh >> 3, h = bh & 7;
  const int d   = kt * 32 + ((lane >> 4) << 3) + (e & 7) + ((e >> 3) << 4);
  const int tok = jt * 16 + (lane & 15);
  kop[i] = kv[(size_t)(b * JTOK + tok) * (2 * INNER) + h * DH + d];
}

// -------- repack V into attention B-operand (tokens = K-dim) ---------------
// Per (b,h): blocks (nt 0..3) x (kt 0..15); elem: tok = kt*32+(lane>>4)*8+(e&7)+(e>>3)*16,
// d = nt*16+(lane&15); value = kv[b*512+tok, 512 + h*64+d]
__global__ void pack_v_kernel(const bf16* __restrict__ kv,
                              bf16* __restrict__ vop, int n) {
  int i = blockIdx.x * blockDim.x + threadIdx.x;
  if (i >= n) return;
  const int e = i & 15, lane = (i >> 4) & 31, blk = i >> 9;
  const int kt = blk & 15;
  const int nt = (blk >> 4) & 3;
  const int bh = blk >> 6;
  const int b = bh >> 3, h = bh & 7;
  const int tok = kt * 32 + ((lane >> 4) << 3) + (e & 7) + ((e >> 3) << 4);
  const int d   = nt * 16 + (lane & 15);
  vop[i] = kv[(size_t)(b * JTOK + tok) * (2 * INNER) + INNER + h * DH + d];
}

// ----------------------------- attention -----------------------------------
// One wave per (b, h, 16-token i-tile). Dense 16x512 scores via WMMA,
// mask with -FLT_MAX (reference semantics incl. uniform fallback), softmax,
// then O = P @ V via WMMA. LDS: S f32 (32KB) + P bf16 (16KB).
__global__ __launch_bounds__(32)
void attn_kernel(const bf16* __restrict__ Q,     // [B*TTXT, INNER]
                 const bf16* __restrict__ Kop,   // packed per (b,h)
                 const bf16* __restrict__ Vop,   // packed per (b,h)
                 const int*  __restrict__ tt,    // [B, TTXT]
                 bf16* __restrict__ O) {         // [B*TTXT, INNER]
  __shared__ float S[16 * 512];
  __shared__ bf16  P[16 * 512];
  __shared__ float rmax_s[16];
  __shared__ float rsum_s[16];

  const int lane  = threadIdx.x;
  const int tile  = blockIdx.x;          // b*HEADS*256 + h*256 + it
  const int it    = tile & 255;
  const int h     = (tile >> 8) & 7;
  const int b     = tile >> 11;
  const int half  = lane >> 4;
  const int l15   = lane & 15;
  const size_t qrow0 = (size_t)(b * TTXT + it * 16);
  const float NEGMAX = -3.402823466e38f;

  // Q A-operand for both dh k-steps (row = l15, K-half = half*8)
  ABReg aq[2];
  {
    const bf16* qp = Q + (qrow0 + l15) * INNER + h * DH + (half << 3);
    aq[0].h[0] = load8(qp);
    aq[0].h[1] = load8(qp + 16);
    aq[1].h[0] = load8(qp + 32);
    aq[1].h[1] = load8(qp + 48);
  }
  int ttr[8];
  {
    const int* tb = tt + b * TTXT + it * 16 + (half << 3);
#pragma unroll
    for (int r = 0; r < 8; ++r) ttr[r] = tb[r];
  }

  float rowmax[8];
#pragma unroll
  for (int r = 0; r < 8; ++r) rowmax[r] = NEGMAX;

  // S = Q K^T over 32 j-tiles
  for (int jt = 0; jt < 32; ++jt) {
    v8f s = {};
#pragma unroll
    for (int ks = 0; ks < 2; ++ks) {
      const bf16* kp = Kop + ((size_t)(((b * HEADS + h) * 32 + jt) * 2 + ks)) * 512
                       + lane * 16;
      ABReg bk;
      bk.h[0] = load8(kp);
      bk.h[1] = load8(kp + 8);
      s = wmma_bf16(aq[ks].v, bk.v, s);
    }
    const int jcol  = jt * 16 + l15;
    const int mtime = (jcol >> 6) + 1;
#pragma unroll
    for (int r = 0; r < 8; ++r) {
      const int m = r + (half << 3);
      const float sv = (ttr[r] == mtime) ? s[r] : NEGMAX;
      S[m * 512 + jcol] = sv;
      rowmax[r] = fmaxf(rowmax[r], sv);
    }
  }
  // row max: reduce across the 16 lanes of each half
#pragma unroll
  for (int off = 1; off < 16; off <<= 1) {
#pragma unroll
    for (int r = 0; r < 8; ++r)
      rowmax[r] = fmaxf(rowmax[r], __shfl_xor(rowmax[r], off));
  }
  if (l15 == 0) {
#pragma unroll
    for (int r = 0; r < 8; ++r) rmax_s[(half << 3) + r] = rowmax[r];
  }
  __syncthreads();

  // exp + row sums; lane handles row l15, its half of the 512 columns
  {
    const int row = l15;
    const float m = rmax_s[row];
    float sum = 0.f;
    const int c0 = half * 256;
    for (int c = 0; c < 256; ++c) {
      const float sv = S[row * 512 + c0 + c];
      const float p = __expf(sv - m);   // uniform 1.0 when whole row masked
      P[row * 512 + c0 + c] = (bf16)p;
      sum += p;
    }
    sum += __shfl_xor(sum, 16);
    if (half == 0) rsum_s[row] = sum;
  }
  __syncthreads();

  // O = P @ V  (K-dim = 512 tokens, N = 64 dh)
  v8f o[4] = {};
  for (int kt = 0; kt < 16; ++kt) {
    ABReg pa;
    const bf16* pp = &P[l15 * 512 + kt * 32 + (half << 3)];
    pa.h[0] = *(const v8bf*)pp;
    pa.h[1] = *(const v8bf*)(pp + 16);
#pragma unroll
    for (int nt = 0; nt < 4; ++nt) {
      const bf16* vp = Vop + ((size_t)(((b * HEADS + h) * 4 + nt) * 16 + kt)) * 512
                       + lane * 16;
      ABReg bv;
      bv.h[0] = load8(vp);
      bv.h[1] = load8(vp + 8);
      o[nt] = wmma_bf16(pa.v, bv.v, o[nt]);
    }
  }
  // normalize by row sum and store
#pragma unroll
  for (int nt = 0; nt < 4; ++nt) {
    const int d = nt * 16 + l15;
#pragma unroll
    for (int r = 0; r < 8; ++r) {
      const int m = r + (half << 3);
      const float val = o[nt][r] / rsum_s[m];
      O[(qrow0 + m) * INNER + h * DH + d] = (bf16)val;
    }
  }
}

// ---------------------------------------------------------------------------
extern "C" void kernel_launch(void* const* d_in, const int* in_sizes, int n_in,
                              void* d_out, int out_size, void* d_ws, size_t ws_size,
                              hipStream_t stream) {
  const float* x        = (const float*)d_in[0];
  const float* media    = (const float*)d_in[1];
  const unsigned char* mloc = (const unsigned char*)d_in[2];  // jax bool = 1B
  const float* ln_scale = (const float*)d_in[3];
  const float* ln_bias  = (const float*)d_in[4];
  const float* w_q      = (const float*)d_in[5];
  const float* w_kv     = (const float*)d_in[6];
  const float* w_out    = (const float*)d_in[7];
  float* out = (float*)d_out;

  char* ws = (char*)d_ws;
  size_t off = 0;
  auto carve = [&](size_t bytes) -> void* {
    void* p = ws + off;
    off += (bytes + 255) & ~(size_t)255;
    return p;
  };
  bf16* xn   = (bf16*)carve((size_t)Bb * TTXT * DIMC * 2);       // LN'd text
  bf16* medb = (bf16*)carve((size_t)Bb * JTOK * DIMC * 2);       // media bf16
  bf16* qb   = (bf16*)carve((size_t)Bb * TTXT * INNER * 2);      // Q (scaled)
  bf16* kvb  = (bf16*)carve((size_t)Bb * JTOK * 2 * INNER * 2);  // K|V
  bf16* wqp  = (bf16*)carve((size_t)DIMC * INNER * 2);
  bf16* wkvp = (bf16*)carve((size_t)DIMC * 2 * INNER * 2);
  bf16* wop  = (bf16*)carve((size_t)INNER * DIMC * 2);
  bf16* kop  = (bf16*)carve((size_t)Bb * HEADS * 64 * 512 * 2);  // K B-operand
  bf16* vop  = (bf16*)carve((size_t)Bb * HEADS * 64 * 512 * 2);  // V B-operand
  bf16* ao   = (bf16*)carve((size_t)Bb * TTXT * INNER * 2);      // attn out
  int*  tt   = (int*)carve((size_t)Bb * TTXT * 4);
  (void)ws_size; (void)in_sizes; (void)n_in; (void)out_size;

  // 1) LayerNorm -> bf16
  ln_to_bf16_kernel<<<Bb * TTXT, 256, 0, stream>>>(x, ln_scale, ln_bias, xn);
  // 2) media -> bf16
  {
    int n = Bb * JTOK * DIMC;
    f32_to_bf16_kernel<<<(n + 255) / 256, 256, 0, stream>>>(media, medb, n);
  }
  // 3) text_time cumsum
  text_time_kernel<<<Bb, 32, 0, stream>>>(mloc, tt);
  // 4) pack weights (fold dh^-0.5 into w_q)
  {
    int n = DIMC * INNER;
    pack_w_kernel<<<(n + 255) / 256, 256, 0, stream>>>(w_q, wqp, DIMC, INNER, 0.125f, n);
  }
  {
    int n = DIMC * 2 * INNER;
    pack_w_kernel<<<(n + 255) / 256, 256, 0, stream>>>(w_kv, wkvp, DIMC, 2 * INNER, 1.0f, n);
  }
  {
    int n = INNER * DIMC;
    pack_w_kernel<<<(n + 255) / 256, 256, 0, stream>>>(w_out, wop, INNER, DIMC, 1.0f, n);
  }
  // 5) Q = xn @ w_q : M=8192 K=1024 N=512 -> (8192/32)*(512/64)=2048 waves
  gemm_bf16_kernel<bf16><<<256, 256, 0, stream>>>(xn, wqp, qb, Bb * TTXT, INNER, DIMC);
  // 6) KV = media @ w_kv : M=1024 K=1024 N=1024 -> (1024/32)*16=512 waves
  gemm_bf16_kernel<bf16><<<64, 256, 0, stream>>>(medb, wkvp, kvb, Bb * JTOK, 2 * INNER, DIMC);
  // 7) repack K / V into attention B-operands
  {
    int n = Bb * HEADS * 64 * 512;
    pack_k_kernel<<<(n + 255) / 256, 256, 0, stream>>>(kvb, kop, n);
    pack_v_kernel<<<(n + 255) / 256, 256, 0, stream>>>(kvb, vop, n);
  }
  // 8) attention: one wave per (b,h,16-row tile)
  attn_kernel<<<Bb * HEADS * (TTXT / 16), 32, 0, stream>>>(qb, kop, vop, tt, ao);
  // 9) out = ao @ w_out : M=8192 K=512 N=1024 -> (8192/32)*16=4096 waves
  gemm_bf16_kernel<float><<<512, 256, 0, stream>>>(ao, wop, out, Bb * TTXT, DIMC, INNER);
}
